// SimpleDNN_50646254354887
// MI455X (gfx1250) — matrix operand order repaired and checked by reading, not compiled, not run on hardware
//
#include <hip/hip_runtime.h>
#include <math.h>

typedef __bf16 bf16;
typedef bf16  v16bf __attribute__((ext_vector_type(16)));
typedef float v8f   __attribute__((ext_vector_type(8)));

#define LN_EPS 1e-5f

// Problem dims
constexpr int Bsz  = 65536;
constexpr int Din  = 1024;
constexpr int Hd   = 512;
constexpr int TM   = 128;           // rows per workgroup
constexpr int NWG  = Bsz / TM;      // 512 workgroups

// LDS strides (bf16 elements), padded for bank-conflict-free access
constexpr int H2S = 514;            // h2 tile stride: (row*257 + k/2) % 64 distinct across rows
constexpr int H1S = 136;            // h1 tile stride: 68 words/row -> +4 banks/row
constexpr int SXS = 40;             // 32-wide K staging: 20 words/row -> conflict-free frag gathers
constexpr int W2S = 136;            // 128-wide K staging for fc2_w chunk

// LDS layout (bytes)
constexpr int OFF_H2    = 0;                          // 128*514*2 = 131584
constexpr int OFF_H1    = OFF_H2 + TM * H2S * 2;      // + 34816
constexpr int OFF_STG   = OFF_H1 + TM * H1S * 2;      // double buf 2*(2*128*40*2)=40960; reused for w2/sms
constexpr int STG_BYTES = 40960;
constexpr int OFF_STATS = OFF_STG + STG_BYTES;        // mu(512) + rs(512) + red(1536)
constexpr int SMEM_BYTES = OFF_STATS + 2560;          // = 209920 bytes (< 320 KB/WGP)

#define WMMA_BF16(a, b, c) \
  __builtin_amdgcn_wmma_f32_16x16x32_bf16(false, (a), false, (b), (short)0, (c), false, false)

// A fragment: 16x32 bf16, row = lane%16; lane<16 holds K {0..7,16..23}, lane>=16 holds {8..15,24..31}
__device__ __forceinline__ v16bf frag_a(const bf16* p, int stride, int row0, int k0, int lane) {
  const int half = lane >> 4;
  const int r = row0 + (lane & 15);
  v16bf a;
#pragma unroll
  for (int e = 0; e < 16; ++e) {
    const int k = k0 + ((e >> 3) << 4) + half * 8 + (e & 7);
    a[e] = p[r * stride + k];
  }
  return a;
}

// B fragment: 32x16 bf16 (K x N), col = lane%16; lane<16 holds K {0..7,16..23}, lane>=16 holds {8..15,24..31}.
// Source is stored row-major [n][k] (i.e. W, since logits = h @ W^T -> B[k][n] = W[n][k]).
__device__ __forceinline__ v16bf frag_b(const bf16* p, int stride, int n0, int k0, int lane) {
  const int half = lane >> 4;
  const int n = n0 + (lane & 15);
  v16bf b;
#pragma unroll
  for (int e = 0; e < 16; ++e) b[e] = p[n * stride + k0 + ((e >> 3) << 4) + half * 8 + (e & 7)];
  return b;
}

// Cooperative load of one K=32 tile pair (x rows + fc1_w rows) into a staging buffer, fp32 -> bf16.
__device__ __forceinline__ void load_k32_tiles(const float* __restrict__ x,
                                               const float* __restrict__ fc1_w,
                                               bf16* __restrict__ stgx, bf16* __restrict__ stgw,
                                               int r0, int kc, int t) {
#pragma unroll
  for (int it = 0; it < 4; ++it) {
    const int idx = t + it * 256;
    const int row = idx >> 3, k4 = (idx & 7) * 4;
    const float4 v = *(const float4*)(x + (size_t)(r0 + row) * Din + kc * 32 + k4);
    bf16* d = &stgx[row * SXS + k4];
    d[0] = (bf16)v.x; d[1] = (bf16)v.y; d[2] = (bf16)v.z; d[3] = (bf16)v.w;
  }
#pragma unroll
  for (int it = 0; it < 4; ++it) {
    const int idx = t + it * 256;
    const int n = idx >> 3, k4 = (idx & 7) * 4;
    float4 v = make_float4(0.f, 0.f, 0.f, 0.f);
    if (n < 120) v = *(const float4*)(fc1_w + (size_t)n * Din + kc * 32 + k4);
    bf16* d = &stgw[n * SXS + k4];
    d[0] = (bf16)v.x; d[1] = (bf16)v.y; d[2] = (bf16)v.z; d[3] = (bf16)v.w;
  }
}

extern "C" __global__ __launch_bounds__(256, 1)
void fused_dnn_tile128(const float* __restrict__ x, const int* __restrict__ labels,
                       const int* __restrict__ domain, const float* __restrict__ fc1_w,
                       const float* __restrict__ fc1_b, const float* __restrict__ fc2_w,
                       const float* __restrict__ fc2_b, const float* __restrict__ ln_g,
                       const float* __restrict__ ln_b, const float* __restrict__ sms,
                       const float* __restrict__ smb, const float* __restrict__ embs,
                       const float* __restrict__ cs_wt, float* __restrict__ out,
                       float* __restrict__ partials) {
  extern __shared__ __align__(16) unsigned char smem[];
  bf16*  h2    = (bf16*)(smem + OFF_H2);
  bf16*  h1    = (bf16*)(smem + OFF_H1);
  bf16*  stg   = (bf16*)(smem + OFF_STG);     // phase1: 2 buffers of (x:5120 + w:5120)
  bf16*  stgw2 = (bf16*)(smem + OFF_STG);     // phase2: fc2_w chunk [128][136]
  float* smsl  = (float*)(smem + OFF_STG);    // phase3: sms [512][10]
  float* lngl  = smsl + Hd * 10;
  float* lnbl  = lngl + Hd;
  float* mu    = (float*)(smem + OFF_STATS);
  float* rs    = mu + TM;
  float* red   = rs + TM;                     // 3*TM floats

  const int t = threadIdx.x;
  const int lane = t & 31, wave = t >> 5;
  const int wm = wave >> 1, wn = wave & 1;    // 4x2 wave grid over 128x128 output tile
  const int half = lane >> 4, l15 = lane & 15;
  const int r0 = blockIdx.x * TM;

  // ---------------- Phase 1: h1 = relu(x @ fc1_w^T + b1), K=1024, double-buffered ----------
  v8f c1[2][4];
#pragma unroll
  for (int i = 0; i < 2; ++i)
#pragma unroll
    for (int j = 0; j < 4; ++j)
#pragma unroll
      for (int e = 0; e < 8; ++e) c1[i][j][e] = 0.f;

  constexpr int BUF = 10240;  // bf16 elems per buffer (x 128*40 + w 128*40)
  load_k32_tiles(x, fc1_w, stg, stg + 5120, r0, 0, t);

  for (int kc = 0; kc < Din / 32; ++kc) {
    __syncthreads();
    const int cur = kc & 1;
    if (kc + 1 < Din / 32) {
      bf16* nb = stg + ((kc + 1) & 1) * BUF;
      load_k32_tiles(x, fc1_w, nb, nb + 5120, r0, kc + 1, t);
    }
    const bf16* sx = stg + cur * BUF;
    const bf16* sw = sx + 5120;
    v16bf a[2], b[4];
#pragma unroll
    for (int i = 0; i < 2; ++i) a[i] = frag_a(sx, SXS, wm * 32 + i * 16, 0, lane);
#pragma unroll
    for (int j = 0; j < 4; ++j) b[j] = frag_b(sw, SXS, wn * 64 + j * 16, 0, lane);
#pragma unroll
    for (int i = 0; i < 2; ++i)
#pragma unroll
      for (int j = 0; j < 4; ++j) c1[i][j] = WMMA_BF16(a[i], b[j], c1[i][j]);
  }

  // bias + relu -> h1 (bf16); cols [120,128) stay exactly zero (W padded 0, bias 0)
#pragma unroll
  for (int i = 0; i < 2; ++i)
#pragma unroll
    for (int j = 0; j < 4; ++j) {
      const int col = wn * 64 + j * 16 + l15;
      const float bb = (col < 120) ? fc1_b[col] : 0.f;
#pragma unroll
      for (int e = 0; e < 8; ++e) {
        const int rl = wm * 32 + i * 16 + half * 8 + e;
        float v = c1[i][j][e] + bb;
        v = (v > 0.f) ? v : 0.f;
        h1[rl * H1S + col] = (bf16)v;
      }
    }
  __syncthreads();

  // ---------------- Phase 2: h2 = relu(h1 @ fc2_w^T + b2), K=128 (padded), N=512 in 4 chunks -
  for (int nc = 0; nc < 4; ++nc) {
#pragma unroll
    for (int it = 0; it < 16; ++it) {
      const int idx = t + it * 256;
      const int row = idx >> 5, k4 = (idx & 31) * 4;
      float4 v = make_float4(0.f, 0.f, 0.f, 0.f);
      if (k4 < 120) v = *(const float4*)(fc2_w + (size_t)(nc * 128 + row) * 120 + k4);
      bf16* d = &stgw2[row * W2S + k4];
      d[0] = (bf16)v.x; d[1] = (bf16)v.y; d[2] = (bf16)v.z; d[3] = (bf16)v.w;
    }
    __syncthreads();

    v8f c2[2][4];
#pragma unroll
    for (int i = 0; i < 2; ++i)
#pragma unroll
      for (int j = 0; j < 4; ++j)
#pragma unroll
        for (int e = 0; e < 8; ++e) c2[i][j][e] = 0.f;

#pragma unroll
    for (int kc = 0; kc < 4; ++kc) {
      v16bf a[2], b[4];
#pragma unroll
      for (int i = 0; i < 2; ++i) a[i] = frag_a(h1, H1S, wm * 32 + i * 16, kc * 32, lane);
#pragma unroll
      for (int j = 0; j < 4; ++j) b[j] = frag_b(stgw2, W2S, wn * 64 + j * 16, kc * 32, lane);
#pragma unroll
      for (int i = 0; i < 2; ++i)
#pragma unroll
        for (int j = 0; j < 4; ++j) c2[i][j] = WMMA_BF16(a[i], b[j], c2[i][j]);
    }

#pragma unroll
    for (int i = 0; i < 2; ++i)
#pragma unroll
      for (int j = 0; j < 4; ++j) {
        const int col = nc * 128 + wn * 64 + j * 16 + l15;
        const float bb = fc2_b[col];
#pragma unroll
        for (int e = 0; e < 8; ++e) {
          const int rl = wm * 32 + i * 16 + half * 8 + e;
          float v = c2[i][j][e] + bb;
          v = (v > 0.f) ? v : 0.f;
          h2[rl * H2S + col] = (bf16)v;
        }
      }
    __syncthreads();
  }

  // ---------------- LayerNorm stats (one thread per row; padded stride -> conflict-free) ----
  if (t < TM) {
    float s = 0.f, s2 = 0.f;
#pragma unroll 8
    for (int c = 0; c < Hd; ++c) {
      const float v = (float)h2[t * H2S + c];
      s += v; s2 += v * v;
    }
    const float m = s * (1.f / (float)Hd);
    mu[t] = m;
    rs[t] = rsqrtf(s2 * (1.f / (float)Hd) - m * m + LN_EPS);
  }

  // stage sms [5,512,2] -> [512][10] and LN params (overwrites phase-2 staging, now free)
  for (int idx = t; idx < Hd * 10; idx += 256) {
    const int h = idx / 10, u = idx - h * 10;
    smsl[idx] = sms[(u >> 1) * (Hd * 2) + h * 2 + (u & 1)];
  }
  for (int idx = t; idx < Hd; idx += 256) { lngl[idx] = ln_g[idx]; lnbl[idx] = ln_b[idx]; }
  __syncthreads();

  // ---------------- Phase 3: normalize + [1x512]@[512x10] + mixture + weighted-CE ----------
  if (t < TM) {
    float acc[10];
#pragma unroll
    for (int u = 0; u < 10; ++u) acc[u] = 0.f;
    const float m = mu[t], r = rs[t];
    for (int h = 0; h < Hd; ++h) {
      const float hv = ((float)h2[t * H2S + h] - m) * r * lngl[h] + lnbl[h];
      const float* w = &smsl[h * 10];
#pragma unroll
      for (int u = 0; u < 10; ++u) acc[u] = fmaf(hv, w[u], acc[u]);
    }
    const int row = r0 + t;
    const float z0c = acc[0] + smb[0];
    const float z1c = acc[1] + smb[1];
    out[1 + (size_t)row * 2 + 0] = z0c;
    out[1 + (size_t)row * 2 + 1] = z1c;

    const int dmn = domain[row];
    float cw[5];
    cw[0] = tanhf(cs_wt[0]);
#pragma unroll
    for (int k = 1; k < 5; ++k) cw[k] = tanhf(embs[dmn * 4 + (k - 1)]);
    float z0s = 0.f, z1s = 0.f;
#pragma unroll
    for (int k = 0; k < 5; ++k) {
      z0s = fmaf(cw[k], acc[k * 2 + 0] + smb[k * 2 + 0], z0s);
      z1s = fmaf(cw[k], acc[k * 2 + 1] + smb[k * 2 + 1], z1s);
    }
    const int y = labels[row];
    const float wl = (y == 1) ? 0.9f : 0.1f;
    auto nll2 = [](float a0, float a1, int yy) {
      const float mx = fmaxf(a0, a1);
      const float lse = mx + logf(expf(a0 - mx) + expf(a1 - mx));
      return lse - ((yy == 1) ? a1 : a0);
    };
    red[t]          = wl * nll2(z0s, z1s, y);   // specific
    red[TM + t]     = wl * nll2(z0c, z1c, y);   // common
    red[2 * TM + t] = wl;
  }
  __syncthreads();
  if (t == 0) {  // deterministic fixed-order per-WG reduction, no atomics
    float a = 0.f, b = 0.f, c = 0.f;
    for (int i = 0; i < TM; ++i) { a += red[i]; b += red[TM + i]; c += red[2 * TM + i]; }
    partials[blockIdx.x * 3 + 0] = a;
    partials[blockIdx.x * 3 + 1] = b;
    partials[blockIdx.x * 3 + 2] = c;
  }
}

extern "C" __global__ void finalize_losses(const float* __restrict__ sms,
                                           const float* __restrict__ partials,
                                           float* __restrict__ out) {
  __shared__ float lred[64];
  const int t = threadIdx.x;
  float sq = 0.f;
  if (t < 50) {  // cps[c][k][j]: 50 length-512 dot products
    const int c = t / 25, rem = t % 25, k = rem / 5, j = rem % 5;
    float dot = 0.f;
    for (int h = 0; h < Hd; ++h)
      dot = fmaf(sms[k * (Hd * 2) + h * 2 + c], sms[j * (Hd * 2) + h * 2 + c], dot);
    const float d = dot - ((k == j) ? 1.f : 0.f);
    sq = d * d;
  }
  lred[t] = sq;
  __syncthreads();
  if (t == 0) {
    float orth = 0.f;
    for (int i = 0; i < 50; ++i) orth += lred[i];
    orth *= (1.f / 50.f);
    float ss = 0.f, sc = 0.f, sw = 0.f;
    for (int i = 0; i < NWG; ++i) {
      ss += partials[i * 3 + 0];
      sc += partials[i * 3 + 1];
      sw += partials[i * 3 + 2];
    }
    const float spec = ss / sw, cls = sc / sw;
    out[0] = cls + spec + orth;                 // loss
    out[1 + (size_t)Bsz * 2 + 0] = cls;         // class_loss
    out[1 + (size_t)Bsz * 2 + 1] = spec;        // specific_loss
    out[1 + (size_t)Bsz * 2 + 2] = orth;        // orth_loss
  }
}

extern "C" void kernel_launch(void* const* d_in, const int* in_sizes, int n_in,
                              void* d_out, int out_size, void* d_ws, size_t ws_size,
                              hipStream_t stream) {
  const float* x      = (const float*)d_in[0];
  const int*   labels = (const int*)d_in[1];
  const int*   domain = (const int*)d_in[2];
  const float* fc1_w  = (const float*)d_in[3];
  const float* fc1_b  = (const float*)d_in[4];
  const float* fc2_w  = (const float*)d_in[5];
  const float* fc2_b  = (const float*)d_in[6];
  const float* ln_g   = (const float*)d_in[7];
  const float* ln_b   = (const float*)d_in[8];
  const float* sms    = (const float*)d_in[9];
  const float* smb    = (const float*)d_in[10];
  const float* embs   = (const float*)d_in[11];
  const float* cswt   = (const float*)d_in[12];
  float* out      = (float*)d_out;
  float* partials = (float*)d_ws;  // 512*3 floats, fully rewritten every call

  hipLaunchKernelGGL(fused_dnn_tile128, dim3(NWG), dim3(256), SMEM_BYTES, stream,
                     x, labels, domain, fc1_w, fc1_b, fc2_w, fc2_b, ln_g, ln_b,
                     sms, smb, embs, cswt, out, partials);
  hipLaunchKernelGGL(finalize_losses, dim3(1), dim3(64), 0, stream, sms, partials, out);
}
